// LinearAttention_43181601194856
// MI455X (gfx1250) — matrix operand order repaired and checked by reading, not compiled, not run on hardware
//
#include <hip/hip_runtime.h>
#include <hip/hip_bf16.h>

typedef __attribute__((ext_vector_type(16))) __bf16 v16bf;
typedef __attribute__((ext_vector_type(8)))  __bf16 v8bf;
typedef __attribute__((ext_vector_type(8)))  float  v8f;

__device__ __forceinline__ v16bf cat16(v8bf lo, v8bf hi) {
  v16bf r;
#pragma unroll
  for (int i = 0; i < 8; ++i) { r[i] = lo[i]; r[i + 8] = hi[i]; }
  return r;
}

// Generic tiled bf16-WMMA GEMM:  C[M,N] = op(A)[M,K] @ B[K,N]
//   TRANS_A=false: A is [M,K] row-major (lda = row stride)
//   TRANS_A=true : logical A[m,k] = Araw[k*lda + m]   (used for pk^T @ pv)
//   B is always logical [K,N] row-major (ldb = row stride)
// EPI: 0 = store bf16
//      1 = v *= mask[row]           -> store bf16   (pv path)
//      2 = v *= mask[row]; swish(v) -> store bf16   (pk path)
//      3 = store f32                                 (final output)
template <typename TA, typename TB, bool TRANS_A, int EPI>
__global__ __launch_bounds__(256)
void gemm_wmma(const TA* __restrict__ A, const TB* __restrict__ B,
               void* __restrict__ C, const float* __restrict__ mask,
               int M, int N, int K, int lda, int ldb, int ldc,
               long long sAz, long long sBz, long long sCz) {
  __shared__ __bf16 lA[128][32];   // [m][k]
  __shared__ __bf16 lB[128][32];   // [n][k]  (B stored transposed in LDS)

  const int tid  = threadIdx.x;
  const int lane = tid & 31;
  const int wid  = tid >> 5;
  const int wM   = wid >> 2;            // 0..1  -> 64 rows each
  const int wN   = wid & 3;             // 0..3  -> 32 cols each
  const int m0   = blockIdx.y * 128;
  const int n0   = blockIdx.x * 128;
  const int z    = blockIdx.z;

  A += (long long)z * sAz;
  B += (long long)z * sBz;

  v8f acc[4][2];
#pragma unroll
  for (int i = 0; i < 4; ++i)
#pragma unroll
    for (int j = 0; j < 2; ++j)
#pragma unroll
      for (int r = 0; r < 8; ++r) acc[i][j][r] = 0.0f;

  const int khalf = (lane >> 4) << 3;   // A frag: 0 or 8
  const int kb16  = (lane >> 4) << 4;   // 0 or 16 (unused; B frag uses khalf)
  const int mrow  = lane & 15;

  for (int k0 = 0; k0 < K; k0 += 32) {
    // ---- stage A tile (128 x 32) into LDS as bf16 [m][k] ----
    if (!TRANS_A) {
      const int row = tid >> 1;                 // 0..127
      const int cb  = (tid & 1) << 4;           // 0 or 16
      const TA* src = A + (long long)(m0 + row) * lda + (k0 + cb);
#pragma unroll
      for (int i = 0; i < 16; ++i) lA[row][cb + i] = (__bf16)(float)src[i];
      if (k0 + 32 < K) __builtin_prefetch(src + 32, 0, 1);
    } else {
      const int kr = tid >> 3;                  // 0..31 (raw row = k)
      const int cb = (tid & 7) << 4;            // m offset 0..112
      const TA* src = A + (long long)(k0 + kr) * lda + (m0 + cb);
#pragma unroll
      for (int i = 0; i < 16; ++i) lA[cb + i][kr] = (__bf16)(float)src[i];
      if (k0 + 32 < K) __builtin_prefetch(src + 32LL * lda, 0, 1);
    }
    // ---- stage B tile (32 x 128) into LDS transposed as bf16 [n][k] ----
    {
      const int kr = tid >> 3;
      const int cb = (tid & 7) << 4;
      const TB* src = B + (long long)(k0 + kr) * ldb + (n0 + cb);
#pragma unroll
      for (int i = 0; i < 16; ++i) lB[cb + i][kr] = (__bf16)(float)src[i];
      if (k0 + 32 < K) __builtin_prefetch(src + 32LL * ldb, 0, 1);
    }
    __syncthreads();

    // ---- fragments per ISA 16-bit WMMA layouts ----
    v16bf afr[4], bfr[2];
#pragma unroll
    for (int i = 0; i < 4; ++i) {
      const __bf16* p = &lA[wM * 64 + i * 16 + mrow][0];
      afr[i] = cat16(*(const v8bf*)(p + khalf), *(const v8bf*)(p + 16 + khalf));
    }
#pragma unroll
    for (int j = 0; j < 2; ++j) {
      const __bf16* p = &lB[wN * 32 + j * 16 + mrow][khalf];
      bfr[j] = cat16(*(const v8bf*)p, *(const v8bf*)(p + 16));
    }
#pragma unroll
    for (int i = 0; i < 4; ++i)
#pragma unroll
      for (int j = 0; j < 2; ++j)
        acc[i][j] = __builtin_amdgcn_wmma_f32_16x16x32_bf16(
            false, afr[i], false, bfr[j], (short)0, acc[i][j], false, false);
    __syncthreads();
  }

  // ---- epilogue: C/D layout = VGPR r -> M = r + 8*(lane>=16), N = lane&15 ----
  const int ncol = lane & 15;
  const int rhi  = (lane >> 4) << 3;
  float*  Cf = (float*)C  + (long long)z * sCz;
  __bf16* Cb = (__bf16*)C + (long long)z * sCz;
#pragma unroll
  for (int i = 0; i < 4; ++i) {
#pragma unroll
    for (int j = 0; j < 2; ++j) {
      const int gmb = m0 + wM * 64 + i * 16 + rhi;
      const int gn  = n0 + wN * 32 + j * 16 + ncol;
#pragma unroll
      for (int r = 0; r < 8; ++r) {
        const int gm = gmb + r;
        float v = acc[i][j][r];
        if (EPI == 1 || EPI == 2) v *= mask[gm];
        if (EPI == 2) v = v / (1.0f + __expf(-v));   // swish
        if (EPI == 3) Cf[(long long)gm * ldc + gn] = v;
        else          Cb[(long long)gm * ldc + gn] = (__bf16)v;
      }
    }
  }
}

extern "C" void kernel_launch(void* const* d_in, const int* in_sizes, int n_in,
                              void* d_out, int out_size, void* d_ws, size_t ws_size,
                              hipStream_t stream) {
  (void)in_sizes; (void)n_in; (void)out_size; (void)ws_size;
  const int Bn = 4, S = 4096, U = 1024;

  const float* q  = (const float*)d_in[0];
  const float* k  = (const float*)d_in[1];
  const float* v  = (const float*)d_in[2];
  const float* mk = (const float*)d_in[3];
  const float* Wk = (const float*)d_in[4];
  const float* Wv = (const float*)d_in[5];
  const float* Wq = (const float*)d_in[6];
  const float* Wo = (const float*)d_in[7];

  const size_t szBSU = (size_t)Bn * S * U;   // elements
  char* ws = (char*)d_ws;
  __bf16* pq  = (__bf16*)(ws);               // 32 MB
  __bf16* pk  = (__bf16*)(ws + szBSU * 2);   // 32 MB
  __bf16* pv  = (__bf16*)(ws + szBSU * 4);   // 32 MB
  __bf16* kvb = (__bf16*)(ws + szBSU * 6);   //  8 MB
  __bf16* ob  = pk;                          // 'out' reuses pk region (dead after kv)

  dim3 blk(256);

  // Projections: [16384,1024] @ [1024,1024]  (f32 in, bf16 out)
  dim3 gp(U / 128, (Bn * S) / 128, 1);
  gemm_wmma<float, float, false, 0><<<gp, blk, 0, stream>>>(
      q, Wq, pq, nullptr, Bn * S, U, U, U, U, U, 0, 0, 0);
  gemm_wmma<float, float, false, 2><<<gp, blk, 0, stream>>>(
      k, Wk, pk, mk, Bn * S, U, U, U, U, U, 0, 0, 0);
  gemm_wmma<float, float, false, 1><<<gp, blk, 0, stream>>>(
      v, Wv, pv, mk, Bn * S, U, U, U, U, U, 0, 0, 0);

  // kv[b] = pk[b]^T @ pv[b] : per batch [1024,1024], K = 4096
  dim3 gkv(U / 128, U / 128, Bn);
  gemm_wmma<__bf16, __bf16, true, 0><<<gkv, blk, 0, stream>>>(
      pk, pv, kvb, nullptr, U, U, S, U, U, U,
      (long long)S * U, (long long)S * U, (long long)U * U);

  // out[b] = pq[b] @ kv[b] : per batch [4096,1024], K = 1024
  dim3 go(U / 128, S / 128, Bn);
  gemm_wmma<__bf16, __bf16, false, 0><<<go, blk, 0, stream>>>(
      pq, kvb, ob, nullptr, S, U, U, U, U, U,
      (long long)S * U, (long long)U * U, (long long)S * U);

  // final = out @ Wo : [16384,1024], f32 output
  dim3 gf(U / 128, (Bn * S) / 128, 1);
  gemm_wmma<__bf16, float, false, 3><<<gf, blk, 0, stream>>>(
      ob, Wo, d_out, nullptr, Bn * S, U, U, U, U, U, 0, 0, 0);
}
